// BasePointSAModule_15599321219412
// MI455X (gfx1250) — matrix operand order repaired and checked by reading, not compiled, not run on hardware
//
#include <hip/hip_runtime.h>
#include <stdint.h>

// ---------------------------------------------------------------------------
// PointNet++ SA module for MI455X (gfx1250, wave32, WMMA).
// GEMM body (23.7 GFLOP) runs on v_wmma_f32_16x16x32_f16. X activations are
// staged block-wide into LDS via global_load_async_to_lds_b128 (ASYNCcnt) so
// every activation byte is fetched from L2/HBM exactly once; all 8 waves of a
// block (covering the full cout extent) consume it via ds_load_b128.
// Workspace requirement: ~102 MB.
// ---------------------------------------------------------------------------

typedef __attribute__((ext_vector_type(16))) _Float16 v16h;
typedef __attribute__((ext_vector_type(8)))  float    v8f;

#define NPTS   16384
#define NB     8
#define NM     1024
#define NK     32
#define NCOLS  (NB * NM * NK)   // 262144
#define RAD2   0.16f

// ---- workspace layout (bytes) ----
#define OFF_IDX  0                         // FPS indices: 8*1024 int
#define OFF_BQ   (32768)                   // ball-query idx: 262144 int
#define OFF_W0   (OFF_BQ + 1048576)        // W0 f16: 64*96
#define OFF_W1   (OFF_W0 + 12288)          // W1 f16: 128*64
#define OFF_W2   (OFF_W1 + 16384)          // W2 f16: 256*128
#define OFF_A    (1179648)                 // region A: 64 MB (g: 48 MB, then h2: 64 MB)
#define OFF_B    (OFF_A + 67108864)        // region B: 32 MB (h1)

// ===========================================================================
// 1) Farthest point sampling: one workgroup per batch, 1024 threads.
//    16 points/thread live entirely in registers; LDS only for argmax.
// ===========================================================================
__global__ __launch_bounds__(1024) void fps_kernel(const float* __restrict__ xyz,
                                                   int* __restrict__ idxOut) {
  __shared__ float candD[1024];
  __shared__ int   candI[1024];
  __shared__ float sC[3];
  __shared__ int   sFar;

  const int b = blockIdx.x;
  const int t = threadIdx.x;
  const float* px = xyz + (size_t)b * NPTS * 3;

  float xs[16], ys[16], zs[16], ds[16];
#pragma unroll
  for (int i = 0; i < 16; ++i) {
    const int p = t + (i << 10);
    xs[i] = px[p * 3 + 0];
    ys[i] = px[p * 3 + 1];
    zs[i] = px[p * 3 + 2];
    ds[i] = 1e10f;
  }
  if (t == 0) { sFar = 0; sC[0] = px[0]; sC[1] = px[1]; sC[2] = px[2]; }
  __syncthreads();

  int far = 0;
  for (int s = 0; s < NM; ++s) {
    if (t == 0) idxOut[b * NM + s] = far;
    const float cx = sC[0], cy = sC[1], cz = sC[2];

    float best = -1.0f; int bestI = 0;
#pragma unroll
    for (int i = 0; i < 16; ++i) {
      const float dx = xs[i] - cx, dy = ys[i] - cy, dz = zs[i] - cz;
      const float d  = dx * dx + dy * dy + dz * dz;
      const float nd = fminf(ds[i], d);
      ds[i] = nd;
      const int p = t + (i << 10);
      if (nd > best || (nd == best && p < bestI)) { best = nd; bestI = p; }
    }
    candD[t] = best; candI[t] = bestI;
    __syncthreads();

    if (t < 32) {                       // wave 0 finishes the reduction
      float bd = candD[t * 32]; int bi = candI[t * 32];
      for (int j = 1; j < 32; ++j) {
        const float d = candD[t * 32 + j]; const int i2 = candI[t * 32 + j];
        if (d > bd || (d == bd && i2 < bi)) { bd = d; bi = i2; }
      }
#pragma unroll
      for (int off = 16; off > 0; off >>= 1) {
        const float d = __shfl_xor(bd, off);
        const int  i2 = __shfl_xor(bi, off);
        if (d > bd || (d == bd && i2 < bi)) { bd = d; bi = i2; }
      }
      if (t == 0) sFar = bi;
    }
    __syncthreads();
    far = sFar;
    if ((far & 1023) == t) {            // owner publishes next centroid
      const int i = far >> 10;
      sC[0] = xs[i]; sC[1] = ys[i]; sC[2] = zs[i];
    }
    __syncthreads();
  }
}

// ===========================================================================
// 2) Gather new_xyz + emit indices (as float) into the output tuple.
// ===========================================================================
__global__ __launch_bounds__(256) void gather_kernel(const float* __restrict__ xyz,
                                                     const int* __restrict__ idx,
                                                     float* __restrict__ out_xyz,
                                                     float* __restrict__ out_idx) {
  const int g = blockIdx.x * blockDim.x + threadIdx.x;
  if (g >= NB * NM) return;
  const int b = g >> 10;
  const int p = idx[g];
  const float* s = xyz + ((size_t)b * NPTS + p) * 3;
  out_xyz[g * 3 + 0] = s[0];
  out_xyz[g * 3 + 1] = s[1];
  out_xyz[g * 3 + 2] = s[2];
  out_idx[g] = (float)p;
}

// ===========================================================================
// 3) Ball query: one wave32 per query point; ordered append via ballot.
// ===========================================================================
__global__ __launch_bounds__(256) void ballquery_kernel(const float* __restrict__ xyz,
                                                        const float* __restrict__ newxyz,
                                                        int* __restrict__ bq) {
  const int wid  = (blockIdx.x * blockDim.x + threadIdx.x) >> 5;
  const int lane = threadIdx.x & 31;
  if (wid >= NB * NM) return;
  const int b = wid >> 10;
  const float* px = xyz + (size_t)b * NPTS * 3;
  const float qx = newxyz[wid * 3 + 0];
  const float qy = newxyz[wid * 3 + 1];
  const float qz = newxyz[wid * 3 + 2];
  int* o = bq + wid * NK;

  int found = 0, first = -1;
  for (int base = 0; base < NPTS && found < NK; base += 32) {
    const int p = base + lane;
    const float dx = px[p * 3 + 0] - qx;
    const float dy = px[p * 3 + 1] - qy;
    const float dz = px[p * 3 + 2] - qz;
    const bool pred = (dx * dx + dy * dy + dz * dz) <= RAD2;
    const unsigned mask = (unsigned)__ballot(pred);
    if (pred) {
      const int pos = found + __popc(mask & ((1u << lane) - 1u));
      if (pos < NK) o[pos] = p;
    }
    if (first < 0 && mask) first = base + __ffs(mask) - 1;
    found += __popc(mask);
  }
  if (found < NK) {                     // pad tail with first in-radius index
    if (first < 0) first = NPTS - 1;
    if (lane >= found && lane < NK) o[lane] = first;
  }
}

// ===========================================================================
// 4) Grouping: build layer-0 activations Xt0[col][96] f16,
//    col = ((b*1024+m)*32+k); ch 0-2 = rel xyz, 3-66 = features, 67-95 = 0.
// ===========================================================================
__global__ __launch_bounds__(256) void group_kernel(const float* __restrict__ xyz,
                                                    const float* __restrict__ feat,
                                                    const float* __restrict__ newxyz,
                                                    const int* __restrict__ bq,
                                                    _Float16* __restrict__ Xt) {
  const int col = blockIdx.x * blockDim.x + threadIdx.x;
  if (col >= NCOLS) return;
  const int q = col >> 5;               // b*1024 + m
  const int b = q >> 10;
  const int p = bq[col];
  const float* s = xyz + ((size_t)b * NPTS + p) * 3;
  _Float16* d = Xt + (size_t)col * 96;
  d[0] = (_Float16)(s[0] - newxyz[q * 3 + 0]);
  d[1] = (_Float16)(s[1] - newxyz[q * 3 + 1]);
  d[2] = (_Float16)(s[2] - newxyz[q * 3 + 2]);
  const float* f = feat + (size_t)b * 64 * NPTS + p;
#pragma unroll 8
  for (int c = 0; c < 64; ++c) d[3 + c] = (_Float16)f[(size_t)c * NPTS];
#pragma unroll
  for (int c = 67; c < 96; ++c) d[c] = (_Float16)0.0f;
}

// ===========================================================================
// 5) Weight conversion f32 -> f16 with K padding.
// ===========================================================================
__global__ __launch_bounds__(256) void wconv_kernel(const float* __restrict__ W,
                                                    _Float16* __restrict__ Wh,
                                                    int cout, int cin, int cinP) {
  const int g = blockIdx.x * blockDim.x + threadIdx.x;
  if (g >= cout * cinP) return;
  const int o = g / cinP, c = g % cinP;
  Wh[g] = (c < cin) ? (_Float16)W[o * cin + c] : (_Float16)0.0f;
}

// ===========================================================================
// 6) WMMA GEMM: Y[cout, col] = W[cout, cinP] x Xt[col, cinP]^T.
//    Block = 8 waves covering the FULL cout extent x (wavesX*32) columns.
//    X strip staged into LDS once per block via global_load_async_to_lds_b128,
//    consumed by all waves via ds_load_b128 (exactly-once HBM traffic).
//    Wave tile 32x32 (2x2 of 16x16), K-steps of 32 via v_wmma_f32_16x16x32_f16.
//    !FINAL: fused affine+ReLU, f16 b128 store into Ynext[col][COUT].
//     FINAL: fused affine+ReLU + max over the 32 samples of one point
//            (shfl_xor butterfly), writes new_features[b][cout][m] f32.
// ===========================================================================
template <int CINP, int COUT, bool FINAL>
__global__ __launch_bounds__(256) void gemm_wmma_kernel(
    const _Float16* __restrict__ Xt,     // [NCOLS][CINP]
    const _Float16* __restrict__ Wh,     // [COUT][CINP]
    const float* __restrict__ bias,
    const float* __restrict__ scale,
    const float* __restrict__ shift,
    _Float16* __restrict__ Ynext,        // [NCOLS][COUT]  (if !FINAL)
    float* __restrict__ outFeat) {       // [NB][256][NM]  (if FINAL)
  constexpr int TILESY  = COUT / 32;     // wave rows  (cout tiles)
  constexpr int WAVESX  = 8 / TILESY;    // wave cols  (column strips)
  constexpr int BLKCOLS = WAVESX * 32;   // columns staged per block
  __shared__ _Float16 tile[BLKCOLS * CINP];   // <= 24 KB

  const int tid  = threadIdx.x;
  const int lane = tid & 31;
  const int ln   = lane & 15;
  const int hi   = lane >> 4;
  const int wib  = tid >> 5;
  const int coutBase = (wib % TILESY) * 32;
  const int wx       = wib / TILESY;
  const int colBlk   = blockIdx.x * BLKCOLS;

  // ---- stage X[colBlk .. colBlk+BLKCOLS) x CINP into LDS (async, 16B/lane) --
  constexpr int CHUNKS = (BLKCOLS * CINP) / (8 * 256);
  const _Float16* gbase = Xt + (size_t)colBlk * CINP;
#pragma unroll
  for (int c = 0; c < CHUNKS; ++c) {
    const int idx = (c * 256 + tid) * 8;
    const unsigned lds_addr = (unsigned)(uintptr_t)(&tile[idx]);
    const _Float16* gp = gbase + idx;
    asm volatile("global_load_async_to_lds_b128 %0, %1, off"
                 :: "v"(lds_addr), "v"(gp) : "memory");
  }
  // warm L2 for the next block's strip while we compute
  __builtin_prefetch(gbase + (size_t)BLKCOLS * CINP, 0, 1);
  asm volatile("s_wait_asynccnt 0x0" ::: "memory");
  __syncthreads();

  v8f acc[2][2];
#pragma unroll
  for (int i = 0; i < 2; ++i)
#pragma unroll
    for (int j = 0; j < 2; ++j)
      acc[i][j] = (v8f){0.f, 0.f, 0.f, 0.f, 0.f, 0.f, 0.f, 0.f};

  const _Float16* wrow0 = Wh + (size_t)(coutBase + ln) * CINP;
  const _Float16* wrow1 = wrow0 + (size_t)16 * CINP;
  const _Float16* lcol0 = tile + (size_t)(wx * 32 + ln) * CINP;
  const _Float16* lcol1 = lcol0 + (size_t)16 * CINP;

#pragma unroll
  for (int kc = 0; kc < CINP; kc += 32) {
    union { v16h v; uint4 q[2]; } a0, a1, b0, b1;
    // A (16x32 f16): slots 0-3 = 16B @ kc+8*hi, slots 4-7 = 16B @ kc+16+8*hi
    a0.q[0] = *(const uint4*)(wrow0 + kc + 8 * hi);
    a0.q[1] = *(const uint4*)(wrow0 + kc + 16 + 8 * hi);
    a1.q[0] = *(const uint4*)(wrow1 + kc + 8 * hi);
    a1.q[1] = *(const uint4*)(wrow1 + kc + 16 + 8 * hi);
    // B (32x16 f16): slots 0-3 = 16B @ kc+8*hi, slots 4-7 = 16B @ kc+16+8*hi (from LDS)
    b0.q[0] = *(const uint4*)(lcol0 + kc + 8 * hi);
    b0.q[1] = *(const uint4*)(lcol0 + kc + 16 + 8 * hi);
    b1.q[0] = *(const uint4*)(lcol1 + kc + 8 * hi);
    b1.q[1] = *(const uint4*)(lcol1 + kc + 16 + 8 * hi);

    acc[0][0] = __builtin_amdgcn_wmma_f32_16x16x32_f16(false, a0.v, false, b0.v, (short)0, acc[0][0], false, false);
    acc[0][1] = __builtin_amdgcn_wmma_f32_16x16x32_f16(false, a0.v, false, b1.v, (short)0, acc[0][1], false, false);
    acc[1][0] = __builtin_amdgcn_wmma_f32_16x16x32_f16(false, a1.v, false, b0.v, (short)0, acc[1][0], false, false);
    acc[1][1] = __builtin_amdgcn_wmma_f32_16x16x32_f16(false, a1.v, false, b1.v, (short)0, acc[1][1], false, false);
  }

  const int colBase = colBlk + wx * 32;
  if (!FINAL) {
#pragma unroll
    for (int i = 0; i < 2; ++i) {
      const int mb = coutBase + 16 * i + 8 * hi;   // 8 consecutive output rows
      float bb[8], ss[8], tt[8];
#pragma unroll
      for (int r = 0; r < 8; ++r) { bb[r] = bias[mb + r]; ss[r] = scale[mb + r]; tt[r] = shift[mb + r]; }
#pragma unroll
      for (int j = 0; j < 2; ++j) {
        const int col = colBase + 16 * j + ln;
        union { uint4 q; _Float16 h[8]; } pk;
#pragma unroll
        for (int r = 0; r < 8; ++r) {
          float y = (acc[i][j][r] + bb[r]) * ss[r] + tt[r];
          pk.h[r] = (_Float16)fmaxf(y, 0.0f);
        }
        *reinterpret_cast<uint4*>(Ynext + (size_t)col * COUT + mb) = pk.q;
      }
    }
  } else {
    // 32 columns of this wave = the 32 samples of one output point.
    const int point = colBase >> 5;
    const int b = point >> 10, m = point & 1023;
#pragma unroll
    for (int i = 0; i < 2; ++i) {
      const int mb = coutBase + 16 * i + 8 * hi;
      float v[8];
#pragma unroll
      for (int r = 0; r < 8; ++r) {
        const float b_ = bias[mb + r], s_ = scale[mb + r], t_ = shift[mb + r];
        const float y0 = fmaxf((acc[i][0][r] + b_) * s_ + t_, 0.0f);
        const float y1 = fmaxf((acc[i][1][r] + b_) * s_ + t_, 0.0f);
        v[r] = fmaxf(y0, y1);
      }
#pragma unroll
      for (int off = 1; off < 16; off <<= 1)
#pragma unroll
        for (int r = 0; r < 8; ++r)
          v[r] = fmaxf(v[r], __shfl_xor(v[r], off));
      if (ln == 0) {
        float* o = outFeat + (size_t)b * 256 * NM + (size_t)mb * NM + m;
#pragma unroll
        for (int r = 0; r < 8; ++r) o[(size_t)r * NM] = v[r];
      }
    }
  }
}

// ===========================================================================
// Launcher
// ===========================================================================
extern "C" void kernel_launch(void* const* d_in, const int* in_sizes, int n_in,
                              void* d_out, int out_size, void* d_ws, size_t ws_size,
                              hipStream_t stream) {
  const float* xyz  = (const float*)d_in[0];   // (8,16384,3)
  const float* feat = (const float*)d_in[1];   // (8,64,16384)
  const float* W0 = (const float*)d_in[2];
  const float* b0 = (const float*)d_in[3];
  const float* s0 = (const float*)d_in[4];
  const float* t0 = (const float*)d_in[5];
  const float* W1 = (const float*)d_in[6];
  const float* b1 = (const float*)d_in[7];
  const float* s1 = (const float*)d_in[8];
  const float* t1 = (const float*)d_in[9];
  const float* W2 = (const float*)d_in[10];
  const float* b2 = (const float*)d_in[11];
  const float* s2 = (const float*)d_in[12];
  const float* t2 = (const float*)d_in[13];

  float* out      = (float*)d_out;
  float* out_xyz  = out;                        // 8*1024*3   = 24576
  float* out_feat = out + 24576;                // 8*256*1024 = 2097152
  float* out_idx  = out + 24576 + 2097152;      // 8*1024     = 8192 (as float)

  char* ws = (char*)d_ws;
  int*      idxFPS = (int*)(ws + OFF_IDX);
  int*      bq     = (int*)(ws + OFF_BQ);
  _Float16* W0h    = (_Float16*)(ws + OFF_W0);
  _Float16* W1h    = (_Float16*)(ws + OFF_W1);
  _Float16* W2h    = (_Float16*)(ws + OFF_W2);
  _Float16* actA   = (_Float16*)(ws + OFF_A);   // g, later h2
  _Float16* actB   = (_Float16*)(ws + OFF_B);   // h1

  fps_kernel<<<NB, 1024, 0, stream>>>(xyz, idxFPS);
  gather_kernel<<<(NB * NM + 255) / 256, 256, 0, stream>>>(xyz, idxFPS, out_xyz, out_idx);
  ballquery_kernel<<<(NB * NM * 32) / 256, 256, 0, stream>>>(xyz, out_xyz, bq);
  group_kernel<<<NCOLS / 256, 256, 0, stream>>>(xyz, feat, out_xyz, bq, actA);

  wconv_kernel<<<(64 * 96 + 255) / 256, 256, 0, stream>>>(W0, W0h, 64, 67, 96);
  wconv_kernel<<<(128 * 64 + 255) / 256, 256, 0, stream>>>(W1, W1h, 128, 64, 64);
  wconv_kernel<<<(256 * 128 + 255) / 256, 256, 0, stream>>>(W2, W2h, 256, 128, 128);

  // blocks cover full cout x (wavesX*32) columns; grid = NCOLS / blockCols
  gemm_wmma_kernel<96, 64, false><<<NCOLS / 128, 256, 0, stream>>>(
      actA, W0h, b0, s0, t0, actB, nullptr);
  gemm_wmma_kernel<64, 128, false><<<NCOLS / 64, 256, 0, stream>>>(
      actB, W1h, b1, s1, t1, actA, nullptr);
  gemm_wmma_kernel<128, 256, true><<<NCOLS / 32, 256, 0, stream>>>(
      actA, W2h, b2, s2, t2, nullptr, out_feat);
}
